// AugmentedLstm_23957327577970
// MI455X (gfx1250) — matrix operand order repaired and checked by reading, not compiled, not run on hardware
//
#include <hip/hip_runtime.h>
#include <hip/hip_bf16.h>
#include <stdint.h>

// Problem constants
#define H_  768
#define B_  32
#define T_  1024
#define D_  768
#define N6  (6*H_)     // 4608
#define N5  (5*H_)     // 3840
#define M_  (B_*T_)    // 32768

// Scan kernel geometry: 48 blocks (one per 16-wide h-slice) x 320 threads (10 waves).
// Wave w: mt = w&1 (two 16-row halves of B_=32), gate = w>>1 (0..4).
#define SCAN_BLOCKS (H_/16)   // 48
#define SCAN_THREADS 320

// LDS layout for the scan kernel (bytes)
#define SM_H_OFF   0                          // 32 x 768 bf16  = 49152
#define SM_WS_OFF  49152                      // 5*16 x 768 bf16 = 122880
#define SM_PS_OFF  (49152 + 122880)           // 5 x 32 x 16 f32 = 10240
#define SM_C_OFF   (49152 + 122880 + 10240)   // 32 x 16 f32     = 2048
#define SM_BYTES   (49152 + 122880 + 10240 + 2048)  // 184320 < 320KB/WGP

typedef __attribute__((ext_vector_type(8)))  __bf16 v8bf;
typedef __attribute__((ext_vector_type(16))) __bf16 v16bf;
typedef __attribute__((ext_vector_type(8)))  float  v8f;
typedef __attribute__((ext_vector_type(4)))  unsigned int v4u;
typedef __attribute__((ext_vector_type(8)))  int v8i;
typedef __attribute__((ext_vector_type(4)))  int v4i;

// ---------- helpers ----------
__device__ __forceinline__ __bf16 f2bf(float x) {
    unsigned u = __builtin_bit_cast(unsigned, x);
    unsigned r = (u + 0x7FFFu + ((u >> 16) & 1u)) >> 16;   // RNE
    unsigned short s = (unsigned short)r;
    return __builtin_bit_cast(__bf16, s);
}

__device__ __forceinline__ v16bf frag_cat(v8bf lo, v8bf hi) {
    return __builtin_shufflevector(lo, hi, 0,1,2,3,4,5,6,7,8,9,10,11,12,13,14,15);
}

// A fragment (16x32 bf16 tile at (m0,k0), A row-major lda):
// lanes 0-15 M=lane, K halves 0..7 / 16..23; lanes 16-31 K +8.
__device__ __forceinline__ v16bf load_a(const __bf16* A, int lda, int m0, int k0, int lane) {
    int m  = m0 + (lane & 15);
    int kh = (lane >> 4) * 8;
    const __bf16* p = A + (size_t)m * lda + (k0 + kh);
    return frag_cat(*(const v8bf*)p, *(const v8bf*)(p + 16));
}

// B fragment for B = W^T (W row-major N x K): lanes 0-15 N=lane hold K 0..7 / 16..23,
// lanes 16-31 K +8.
__device__ __forceinline__ v16bf load_bT(const __bf16* W, int ldw, int k0, int n0, int lane) {
    int n  = n0 + (lane & 15);
    int kh = (lane >> 4) * 8;
    const __bf16* p = W + (size_t)n * ldw + (k0 + kh);
    return frag_cat(*(const v8bf*)p, *(const v8bf*)(p + 16));
}

__device__ __forceinline__ v8f wmma_bf16(v16bf a, v16bf b, v8f c) {
    return __builtin_amdgcn_wmma_f32_16x16x32_bf16(false, a, false, b, (short)0, c, false, false);
}

__device__ __forceinline__ float sigmoidf_(float x) {
    return 1.0f / (1.0f + __expf(-x));
}

// ---------- Tensor Data Mover: 2-D tile (rows x cols of bf16) global -> LDS ----------
// D# per CDNA5 ISA. gptr = tile start; LDS gets rows packed.
// This toolchain exposes the 6-arg builtin: (g0 u32x4, g1 i32x8, g2 i32x4, g3 i32x4,
// extra i32x8, cpol i32).
__device__ __forceinline__ void tdm_load_bf16_2d(const void* gptr, void* lptr,
                                                 unsigned tile_cols, unsigned tile_rows,
                                                 unsigned tensor_d0, unsigned tensor_d1,
                                                 unsigned row_stride_elems) {
    unsigned long long ga = (unsigned long long)(uintptr_t)gptr;
    // Flat LDS addresses carry the LDS byte offset in the low 32 bits (aperture in high bits).
    unsigned la = (unsigned)(uintptr_t)lptr;
    v4u g0;
    g0[0] = 1u;                                            // count=1, user mode, no gather
    g0[1] = la;                                            // lds_addr
    g0[2] = (unsigned)(ga & 0xFFFFFFFFu);                  // global_addr[31:0]
    g0[3] = (unsigned)((ga >> 32) & 0x01FFFFFFu) | 0x80000000u; // addr[56:32] | type=2
    v8i g1;
    g1[0] = 0x10000;                                       // data_size=1 (2B), wg_mask=0
    g1[1] = (int)((tensor_d0 & 0xFFFFu) << 16);            // tensor_dim0[15:0]
    g1[2] = (int)(((tensor_d0 >> 16) & 0xFFFFu) | ((tensor_d1 & 0xFFFFu) << 16));
    g1[3] = (int)((((tensor_d1 >> 16) & 0xFFFFu)) | ((tile_cols & 0xFFFFu) << 16));
    g1[4] = (int)(tile_rows & 0xFFFFu);                    // tile_dim1 (tile_dim2 = 0)
    g1[5] = (int)row_stride_elems;                         // tensor_dim0_stride[31:0]
    g1[6] = 0;                                             // stride0[47:32], dim1_stride lo
    g1[7] = 0;
    v4i z4 = {0, 0, 0, 0};                                 // groups 2/3 unused (<=2D)
    v8i z8 = {0, 0, 0, 0, 0, 0, 0, 0};
    __builtin_amdgcn_tensor_load_to_lds(g0, g1, z4, z4, z8, 0);
}

// ---------- fp32 -> bf16 conversion ----------
__global__ void cvt_bf16_kernel(const float* __restrict__ src, __bf16* __restrict__ dst, int n) {
    for (int i = blockIdx.x * blockDim.x + threadIdx.x; i < n; i += gridDim.x * blockDim.x)
        dst[i] = f2bf(src[i]);
}

// ---------- state init (h=0, barrier=0) ----------
__global__ void init_state_kernel(__bf16* hbf, unsigned* bar) {
    int i = blockIdx.x * blockDim.x + threadIdx.x;
    if (i < B_ * H_) hbf[i] = __builtin_bit_cast(__bf16, (unsigned short)0);
    if (i < 8) bar[i] = 0u;
}

// ---------- Phase 1: proj_in = X @ W_in^T + b_in ----------
// Each wave computes a 32(M) x 64(N) tile: 8 accumulators, A frags reused x4, B x2.
__global__ void proj_gemm_kernel(const __bf16* __restrict__ X,     // M_ x D_
                                 const __bf16* __restrict__ Win,   // N6 x D_
                                 const float*  __restrict__ b_in,  // N6
                                 float*        __restrict__ proj)  // M_ x N6
{
    const int lane = threadIdx.x & 31;
    const int wave = (blockIdx.x * blockDim.x + threadIdx.x) >> 5;
    const int ntN  = N6 / 64;              // 72
    const int mt   = wave / ntN;           // 0..1023
    const int nt   = wave % ntN;
    if (mt >= M_ / 32) return;             // uniform per wave
    const int m0 = mt * 32, n0 = nt * 64;

    v8f acc[2][4] = {};
    for (int k0 = 0; k0 < D_; k0 += 32) {
        v16bf a0 = load_a(X, D_, m0,      k0, lane);
        v16bf a1 = load_a(X, D_, m0 + 16, k0, lane);
#pragma unroll
        for (int j = 0; j < 4; ++j) {
            v16bf b = load_bT(Win, D_, k0, n0 + 16 * j, lane);
            acc[0][j] = wmma_bf16(a0, b, acc[0][j]);
            acc[1][j] = wmma_bf16(a1, b, acc[1][j]);
        }
    }

    const int nl = lane & 15;
    const int hf = lane >> 4;
#pragma unroll
    for (int i = 0; i < 2; ++i) {
#pragma unroll
        for (int j = 0; j < 4; ++j) {
            int n = n0 + 16 * j + nl;
            float bias = b_in[n];
#pragma unroll
            for (int r = 0; r < 8; ++r) {
                int m = m0 + 16 * i + r + 8 * hf;
                proj[(size_t)m * N6 + n] = acc[i][j][r] + bias;
            }
        }
    }
}

// ---------- device-wide barrier ----------
__device__ __forceinline__ void grid_barrier(unsigned* cnt, unsigned* gen, unsigned nblocks) {
    __threadfence();
    __syncthreads();
    if (threadIdx.x == 0) {
        unsigned g = __atomic_load_n(gen, __ATOMIC_RELAXED);
        unsigned arrived = atomicAdd(cnt, 1u);
        if (arrived == nblocks - 1u) {
            __atomic_store_n(cnt, 0u, __ATOMIC_RELAXED);
            __threadfence();
            atomicAdd(gen, 1u);
        } else {
            while (__atomic_load_n(gen, __ATOMIC_RELAXED) == g)
                __builtin_amdgcn_s_sleep(1);
        }
        __threadfence();
    }
    __syncthreads();
}

// ---------- Phase 2: persistent LSTM scan ----------
// Block b owns output columns hh0..hh0+15 for ALL 5 gates (ps stays in LDS, c is
// LDS-resident). One device barrier per step (h all-to-all). W_s gate panels are
// TDM-preloaded once; h is TDM-loaded into LDS each step.
__global__ void lstm_scan_kernel(const float*  __restrict__ proj,    // M_ x N6
                                 const __bf16* __restrict__ Ws,      // N5 x H_
                                 const float*  __restrict__ b_s,     // N5
                                 const int*    __restrict__ lengths, // B_
                                 __bf16*       __restrict__ hbf,     // B_ x H_ (global)
                                 unsigned*     __restrict__ bar,
                                 float*        __restrict__ out)     // B_ x T_ x H_
{
    extern __shared__ char smem[];
    __bf16* sh_h  = (__bf16*)(smem + SM_H_OFF);    // 32 x 768
    __bf16* sh_ws = (__bf16*)(smem + SM_WS_OFF);   // (5*16) x 768
    float*  sh_ps = (float*) (smem + SM_PS_OFF);   // 5 x 32 x 16
    float*  sh_c  = (float*) (smem + SM_C_OFF);    // 32 x 16

    const int tid  = threadIdx.x;
    const int lane = tid & 31;
    const int wid  = tid >> 5;            // 0..9
    const int mt   = wid & 1;             // M half
    const int gsel = wid >> 1;            // gate 0..4
    const int hh0  = blockIdx.x * 16;     // column slice
    const int m0   = mt * 16;
    const int nl   = lane & 15;
    const int hf   = lane >> 4;

    unsigned* cnt = bar;
    unsigned* gen = bar + 1;

    // One-time: TDM the 5 W_s gate panels (16 rows x 768 cols each) into LDS.
    if (mt == 0) {   // waves 0,2,4,6,8 -> one gate tile each
        const __bf16* gsrc = Ws + (size_t)(gsel * H_ + hh0) * H_;
        tdm_load_bf16_2d(gsrc, sh_ws + (size_t)gsel * 16 * H_,
                         /*tile_cols=*/H_, /*tile_rows=*/16,
                         /*tensor_d0=*/H_, /*tensor_d1=*/N5,
                         /*row_stride=*/H_);
        __builtin_amdgcn_s_wait_tensorcnt(0);
    }
    for (int e = tid; e < B_ * 16; e += SCAN_THREADS) sh_c[e] = 0.0f;  // c0 = 0
    const float bsv = b_s[gsel * H_ + hh0 + nl];    // ps bias, per-lane constant
    __syncthreads();

    for (int t = 0; t < T_; ++t) {
        // ---- stage h (32 x 768 bf16, 48KB) into LDS via TDM (wave 0)
        if (wid == 0) {
            tdm_load_bf16_2d(hbf, sh_h,
                             /*tile_cols=*/B_ * H_, /*tile_rows=*/1,
                             /*tensor_d0=*/B_ * H_, /*tensor_d1=*/1,
                             /*row_stride=*/B_ * H_);
            __builtin_amdgcn_s_wait_tensorcnt(0);
        }
        __syncthreads();

        // ---- step GEMM tile: ps[gate][m0..m0+16][hh0..hh0+16], K = 768
        v8f acc = {};
        for (int k0 = 0; k0 < H_; k0 += 32) {
            v16bf a = load_a(sh_h, H_, m0, k0, lane);                    // ds_load_b128
            v16bf b = load_bT(sh_ws, H_, k0, gsel * 16, lane);           // ds_load_b128
            acc = wmma_bf16(a, b, acc);
        }
#pragma unroll
        for (int r = 0; r < 8; ++r) {
            int m = m0 + r + 8 * hf;                 // 0..31
            sh_ps[gsel * (B_ * 16) + m * 16 + nl] = acc[r] + bsv;
        }
        __syncthreads();

        // ---- gates + state update + masked output (block-local columns)
        for (int e = tid; e < B_ * 16; e += SCAN_THREADS) {
            int brow = e >> 4;
            int col  = e & 15;
            int hh   = hh0 + col;
            bool msk = t < lengths[brow];
            const float* pib = proj + ((size_t)brow * T_ + t) * N6;

            float i_ = sigmoidf_(pib[0*H_ + hh] + sh_ps[0*(B_*16) + e]);
            float f_ = sigmoidf_(pib[1*H_ + hh] + sh_ps[1*(B_*16) + e]);
            float g_ = tanhf    (pib[2*H_ + hh] + sh_ps[2*(B_*16) + e]);
            float o_ = sigmoidf_(pib[3*H_ + hh] + sh_ps[3*(B_*16) + e]);
            float c_old = sh_c[e];
            float c_new = i_ * g_ + f_ * c_old;
            float ot    = o_ * tanhf(c_new);
            float hw    = sigmoidf_(pib[4*H_ + hh] + sh_ps[4*(B_*16) + e]);
            float y     = hw * ot + (1.0f - hw) * pib[5*H_ + hh];

            out[((size_t)brow * T_ + t) * H_ + hh] = msk ? y : 0.0f;
            if (msk) {
                sh_c[e] = c_new;
                hbf[brow * H_ + hh] = f2bf(y);       // global: next-step GEMM input
            }
        }

        grid_barrier(cnt, gen, SCAN_BLOCKS);         // h fully updated everywhere
    }
}

// ---------- host side ----------
extern "C" void kernel_launch(void* const* d_in, const int* in_sizes, int n_in,
                              void* d_out, int out_size, void* d_ws, size_t ws_size,
                              hipStream_t stream) {
    const float* inputs  = (const float*)d_in[0];   // B,T,D
    const float* W_in    = (const float*)d_in[1];   // 6H,D
    const float* b_in    = (const float*)d_in[2];   // 6H
    const float* W_s     = (const float*)d_in[3];   // 5H,H
    const float* b_s     = (const float*)d_in[4];   // 5H
    const int*   lengths = (const int*)d_in[5];     // B
    float* out = (float*)d_out;

    char* ws = (char*)d_ws;
    size_t off = 0;
    auto carve = [&](size_t bytes) -> void* {
        off = (off + 255) & ~(size_t)255;
        void* p = ws + off;
        off += bytes;
        return p;
    };
    __bf16*   Xbf   = (__bf16*)carve((size_t)M_ * D_ * 2);   // 50.3 MB
    __bf16*   Winbf = (__bf16*)carve((size_t)N6 * D_ * 2);   //  7.1 MB
    __bf16*   Wsbf  = (__bf16*)carve((size_t)N5 * H_ * 2);   //  5.9 MB
    float*    proj  = (float*) carve((size_t)M_ * N6 * 4);   // 604  MB
    __bf16*   hbf   = (__bf16*)carve((size_t)B_ * H_ * 2);
    unsigned* bar   = (unsigned*)carve(256);
    (void)ws_size; (void)in_sizes; (void)n_in; (void)out_size;

    // 1) convert to bf16
    cvt_bf16_kernel<<<4096, 256, 0, stream>>>(inputs, Xbf, M_ * D_);
    cvt_bf16_kernel<<<2048, 256, 0, stream>>>(W_in, Winbf, N6 * D_);
    cvt_bf16_kernel<<<2048, 256, 0, stream>>>(W_s, Wsbf, N5 * H_);

    // 2) init recurrent state + barrier
    init_state_kernel<<<(B_ * H_ + 255) / 256, 256, 0, stream>>>(hbf, bar);

    // 3) input projection GEMM: 1024 m-tiles x 72 n-strips waves, 8 waves/block
    proj_gemm_kernel<<<(M_ / 32) * (N6 / 64) / 8, 256, 0, stream>>>(Xbf, Winbf, b_in, proj);

    // 4) persistent recurrent scan: 48 blocks x 320 threads, 184KB LDS each
    lstm_scan_kernel<<<SCAN_BLOCKS, SCAN_THREADS, SM_BYTES, stream>>>(
        proj, Wsbf, b_s, lengths, hbf, bar, out);
}
